// AttnResBlockUp_30339648979473
// MI455X (gfx1250) — matrix operand order repaired and checked by reading, not compiled, not run on hardware
//
#include <hip/hip_runtime.h>

// ---------------------------------------------------------------------------
// Types / constants
// ---------------------------------------------------------------------------
typedef unsigned short bf16_t;                                   // raw bf16 bits
typedef __attribute__((ext_vector_type(16))) __bf16 v16bf;
typedef __attribute__((ext_vector_type(8)))  __bf16 v8bf;
typedef __attribute__((ext_vector_type(8)))  float  v8f;

constexpr int cB    = 16;
constexpr int cIN   = 512;
constexpr int cOUT  = 256;
constexpr int cGD   = 256;
constexpr int cTD   = 256;
constexpr int cL    = 32;
constexpr int cP1   = 32 * 32;     // 1024
constexpr int cP2   = 64 * 64;     // 4096
constexpr int cCOND = cGD + cTD;   // 512

// ---------------------------------------------------------------------------
// bf16 helpers (round-to-nearest-even)
// ---------------------------------------------------------------------------
__device__ __forceinline__ bf16_t f2b(float f) {
  unsigned u = __builtin_bit_cast(unsigned, f);
  unsigned r = u + 0x7FFFu + ((u >> 16) & 1u);
  return (bf16_t)(r >> 16);
}
__device__ __forceinline__ float b2f(bf16_t h) {
  unsigned u = ((unsigned)h) << 16;
  return __builtin_bit_cast(float, u);
}
__device__ __forceinline__ __bf16 f2h(float f)  { return __builtin_bit_cast(__bf16, f2b(f)); }
__device__ __forceinline__ float  h2f(__bf16 h) { return b2f(__builtin_bit_cast(bf16_t, h)); }

// ---------------------------------------------------------------------------
// WMMA wrapper + fragment index maps used here (contiguous k; wave32, 16-bit operands)
//   A (16x32):  lane l holds row M=l&15;  element e -> K = (l>>4)*16 + e
//   B (32x16):  lane l holds col N=l&15;  element e -> K = (l>>4)*16 + e
//   C/D (16x16 f32): lane l col N=l&15; vgpr r -> row M = r + 8*(l>>4)
// ---------------------------------------------------------------------------
__device__ __forceinline__ v8f wmma_bf16(v16bf a, v16bf b, v8f c) {
  return __builtin_amdgcn_wmma_f32_16x16x32_bf16(false, a, false, b, (short)0, c,
                                                 false, false);
}
__device__ __forceinline__ int a_kidx(int lane, int e) {
  return ((lane >> 4) << 4) + e;
}

__device__ __forceinline__ v16bf ldv16(const bf16_t* p) {
  return *reinterpret_cast<const v16bf*>(p);
}
__device__ __forceinline__ v8bf ldv8(const bf16_t* p) {
  return *reinterpret_cast<const v8bf*>(p);
}
__device__ __forceinline__ v16bf join8(v8bf lo, v8bf hi) {
  return __builtin_shufflevector(lo, hi, 0, 1, 2, 3, 4, 5, 6, 7,
                                 8, 9, 10, 11, 12, 13, 14, 15);
}
__device__ __forceinline__ void stv8(bf16_t* p, v8bf v) {
  *reinterpret_cast<v8bf*>(p) = v;
}
__device__ __forceinline__ v8bf pack_acc(v8f acc) {
  v8bf d;
#pragma unroll
  for (int r = 0; r < 8; ++r) d[r] = f2h(acc[r]);
  return d;
}

// ---------------------------------------------------------------------------
// K0: pack a weight matrix into fragment-major bf16:
//   dst[tap][(mt*Kt + kt)*32 + lane][16], element (m,k) read from
//   src[m*rowPitch + (colOff+k)*kStride + tap]
// ---------------------------------------------------------------------------
__global__ void k_pack_a(const float* __restrict__ src, bf16_t* __restrict__ dst,
                         int M, int K, int rowPitch, int colOff, int kStride) {
  int tap = blockIdx.y;
  int idx = blockIdx.x * 256 + threadIdx.x;     // (mt*Kt + kt)*32 + lane
  int total = (M / 16) * (K / 32) * 32;
  if (idx >= total) return;
  int lane = idx & 31;
  int t = idx >> 5;
  int Kt = K >> 5;
  int kt = t % Kt, mt = t / Kt;
  int m = mt * 16 + (lane & 15);
  bf16_t* o = dst + ((size_t)tap * total + idx) * 16;
#pragma unroll
  for (int e = 0; e < 16; ++e) {
    int k = kt * 32 + a_kidx(lane, e);
    o[e] = f2b(src[(size_t)m * rowPitch + (size_t)(colOff + k) * kStride + tap]);
  }
}

// ---------------------------------------------------------------------------
// K1: NCHW f32 -> channels-last bf16   dst[(b*N+p)*C + c]
// ---------------------------------------------------------------------------
__global__ void k_cast_cl(const float* __restrict__ src, bf16_t* __restrict__ dst,
                          int C, int N) {
  int i = blockIdx.x * 256 + threadIdx.x;
  int tot = cB * C * N;
  if (i >= tot) return;
  int c = i % C;
  int t = i / C;
  int p = t % N, b = t / N;
  dst[i] = f2b(src[((size_t)b * C + c) * N + p]);
}

// ---------------------------------------------------------------------------
// K2: BatchNorm stats (mean, rsqrt(var+eps)) over channels-last bf16
// ---------------------------------------------------------------------------
__global__ void k_bn_stats_cl(const bf16_t* __restrict__ act, float* __restrict__ mean,
                              float* __restrict__ rstd, int C, int N) {
  int c = blockIdx.x;
  int tot = cB * N;
  float s = 0.f, ss = 0.f;
  for (int i = threadIdx.x; i < tot; i += 256) {
    float v = b2f(act[(size_t)i * C + c]);
    s += v; ss += v * v;
  }
  __shared__ float sh0[256], sh1[256];
  sh0[threadIdx.x] = s; sh1[threadIdx.x] = ss;
  __syncthreads();
  for (int st = 128; st > 0; st >>= 1) {
    if (threadIdx.x < st) {
      sh0[threadIdx.x] += sh0[threadIdx.x + st];
      sh1[threadIdx.x] += sh1[threadIdx.x + st];
    }
    __syncthreads();
  }
  if (threadIdx.x == 0) {
    float m = sh0[0] / tot;
    float v = sh1[0] / tot - m * m;
    mean[c] = m;
    rstd[c] = rsqrtf(v + 1e-5f);
  }
}

// ---------------------------------------------------------------------------
// K3: L2-normalize words over channels; emit both layouts:
//     wlc[b][TD][L] (A-operand of ctx GEMM), wcl[b][L][TD] (B-operand of sim)
// ---------------------------------------------------------------------------
__global__ void k_words_norm(const float* __restrict__ words, bf16_t* __restrict__ wlc,
                             bf16_t* __restrict__ wcl) {
  int i = blockIdx.x * 256 + threadIdx.x;   // b*L + l
  if (i >= cB * cL) return;
  int b = i / cL, l = i - b * cL;
  const float* src = words + (size_t)b * cTD * cL + l;
  float s = 0.f;
  for (int t = 0; t < cTD; ++t) { float v = src[(size_t)t * cL]; s += v * v; }
  float inv = 1.f / fmaxf(sqrtf(s), 1e-12f);
  for (int t = 0; t < cTD; ++t) {
    bf16_t v = f2b(src[(size_t)t * cL] * inv);
    wlc[((size_t)b * cTD + t) * cL + l] = v;
    wcl[((size_t)b * cL + l) * cTD + t] = v;
  }
}

// ---------------------------------------------------------------------------
// K4: embed GEMM  out[b][p][m] = sum_k Apk(m,k) * act[b][p][k]   (all CL)
// ---------------------------------------------------------------------------
__global__ void k_gemm_embed(const bf16_t* __restrict__ Apk, const bf16_t* __restrict__ act,
                             bf16_t* __restrict__ out, int M, int K, int N) {
  int b = blockIdx.z, mt = blockIdx.y, nt = blockIdx.x * 16;
  int lane = threadIdx.x, m = lane & 15, khalf = lane >> 4;
  int Kt = K >> 5;
  const bf16_t* actb = act + ((size_t)b * N + nt + m) * K + (khalf << 4);
  const v16bf* ap = reinterpret_cast<const v16bf*>(Apk) + (size_t)mt * Kt * 32 + lane;
  v8f acc = {};
  for (int kt = 0; kt < Kt; ++kt) {
    if (kt + 1 < Kt) __builtin_prefetch(actb + (size_t)(kt + 1) * 32, 0, 1);
    v16bf a = ap[(size_t)kt * 32];
    v16bf bb = ldv16(actb + (size_t)kt * 32);
    acc = wmma_bf16(a, bb, acc);
  }
  stv8(out + ((size_t)b * N + nt + m) * M + mt * 16 + (khalf << 3), pack_acc(acc));
}

// ---------------------------------------------------------------------------
// K5: L2-normalize each pixel row over TD channels (channels-last, in place)
// ---------------------------------------------------------------------------
__global__ void k_colnorm_cl(bf16_t* __restrict__ img, int N) {
  int i = blockIdx.x * 256 + threadIdx.x;   // b*N + p
  if (i >= cB * N) return;
  bf16_t* row = img + (size_t)i * cTD;
  float s = 0.f;
  for (int t = 0; t < cTD; ++t) { float v = b2f(row[t]); s += v * v; }
  float inv = 1.f / fmaxf(sqrtf(s), 1e-12f);
  for (int t = 0; t < cTD; ++t) row[t] = f2b(b2f(row[t]) * inv);
}

// ---------------------------------------------------------------------------
// K6: context = softmax(imgn . wordsn, mask) @ wordsn^T   (two WMMA GEMMs)
//     one wave per 16 pixels; attn staged through LDS
// ---------------------------------------------------------------------------
__global__ void k_context(const bf16_t* __restrict__ imgn,   // [b][N][TD] CL
                          const bf16_t* __restrict__ wcl,    // [b][L][TD]
                          const bf16_t* __restrict__ wlc,    // [b][TD][L]
                          const unsigned char* __restrict__ mask,
                          bf16_t* __restrict__ ctx,          // [b][N][TD] CL
                          int N) {
  int b = blockIdx.y, pt = blockIdx.x * 16;
  int lane = threadIdx.x, m = lane & 15, khalf = lane >> 4;
  __shared__ float  sim[16][32];
  __shared__ bf16_t attnP[16][32];   // [p][l], rows contiguous -> v16bf B-fragments

  const bf16_t* arow  = imgn + ((size_t)b * N + pt + m) * cTD + (khalf << 4);
  const bf16_t* b0row = wcl + ((size_t)b * cL + m) * cTD + (khalf << 4);
  const bf16_t* b1row = wcl + ((size_t)b * cL + 16 + m) * cTD + (khalf << 4);

  v8f s0 = {}, s1 = {};
  for (int kt = 0; kt < (cTD >> 5); ++kt) {
    v16bf a = ldv16(arow + kt * 32);
    s0 = wmma_bf16(a, ldv16(b0row + kt * 32), s0);
    s1 = wmma_bf16(a, ldv16(b1row + kt * 32), s1);
  }
#pragma unroll
  for (int r = 0; r < 8; ++r) {
    int row = r + (khalf << 3);
    sim[row][m]      = s0[r];
    sim[row][16 + m] = s1[r];
  }
  __syncthreads();

  if (lane < 16) {
    const unsigned char* mk = mask + b * cL;
    float v[32], mx = -3.4e38f;
#pragma unroll
    for (int l = 0; l < cL; ++l) {
      float s = mk[l] ? -3.4e38f : sim[lane][l];
      v[l] = s; mx = fmaxf(mx, s);
    }
    float sum = 0.f;
#pragma unroll
    for (int l = 0; l < cL; ++l) { float e = expf(v[l] - mx); v[l] = e; sum += e; }
    float inv = 1.f / fmaxf(sum, 1e-20f);
#pragma unroll
    for (int l = 0; l < cL; ++l) attnP[lane][l] = f2b(v[l] * inv);
  }
  __syncthreads();

  // ctx[p][c] = sum_l wordsn[c][l] * attn[p][l]   (K = 32 = L)
  v16bf batt = ldv16(&attnP[m][khalf << 4]);
  const bf16_t* wlc_b = wlc + (size_t)b * cTD * cL;
  bf16_t* crow = ctx + ((size_t)b * N + pt + m) * cTD;
  for (int ct = 0; ct < cTD / 16; ++ct) {
    const bf16_t* ar = wlc_b + (size_t)(ct * 16 + m) * cL + (khalf << 4);
    v16bf a = ldv16(ar);
    v8f acc = {};
    acc = wmma_bf16(a, batt, acc);
    stv8(crow + ct * 16 + (khalf << 3), pack_acc(acc));
  }
}

// ---------------------------------------------------------------------------
// K7: per-batch FiLM base  base_g[b][o] = W_g[o,:GD] . gcond[b] + b_g[o]
// ---------------------------------------------------------------------------
__global__ void k_cond_base(const float* __restrict__ wg, const float* __restrict__ bg,
                            const float* __restrict__ wb, const float* __restrict__ bb,
                            const float* __restrict__ gcond,
                            float* __restrict__ baseg, float* __restrict__ baseb, int Cout) {
  int i = blockIdx.x * 256 + threadIdx.x;  // b*Cout + o
  if (i >= cB * Cout) return;
  int b = i / Cout, o = i - b * Cout;
  float sg = bg[o], sb = bb[o];
  const float* gc = gcond + (size_t)b * cGD;
  for (int g = 0; g < cGD; ++g) {
    float c = gc[g];
    sg += wg[(size_t)o * cCOND + g] * c;
    sb += wb[(size_t)o * cCOND + g] * c;
  }
  baseg[i] = sg;
  baseb[i] = sb;
}

// ---------------------------------------------------------------------------
// K8: FiLM (gamma/beta WMMA GEMMs over ctx) + BN apply + ReLU -> bf16 CL
// ---------------------------------------------------------------------------
__global__ void k_film(const bf16_t* __restrict__ Gpk, const bf16_t* __restrict__ Bpk,
                       const float* __restrict__ baseg, const float* __restrict__ baseb,
                       const bf16_t* __restrict__ ctx, const bf16_t* __restrict__ act,
                       const float* __restrict__ mean, const float* __restrict__ rstd,
                       const float* __restrict__ bnw, const float* __restrict__ bnb,
                       bf16_t* __restrict__ outp, int Cout, int N) {
  int b = blockIdx.z, ot = blockIdx.y, pt = blockIdx.x * 16;
  int lane = threadIdx.x, m = lane & 15, khalf = lane >> 4;
  const int Kt = cTD >> 5;
  const v16bf* gp = reinterpret_cast<const v16bf*>(Gpk) + (size_t)ot * Kt * 32 + lane;
  const v16bf* bp = reinterpret_cast<const v16bf*>(Bpk) + (size_t)ot * Kt * 32 + lane;
  const bf16_t* cb = ctx + ((size_t)b * N + pt + m) * cTD + (khalf << 4);
  v8f ag = {}, ab = {};
  for (int kt = 0; kt < Kt; ++kt) {
    v16bf fc = ldv16(cb + (size_t)kt * 32);
    ag = wmma_bf16(gp[(size_t)kt * 32], fc, ag);
    ab = wmma_bf16(bp[(size_t)kt * 32], fc, ab);
  }
  int p = pt + m;
  int o0 = ot * 16 + (khalf << 3);
  v8bf av = ldv8(act + ((size_t)b * N + p) * Cout + o0);
  v8bf dv;
#pragma unroll
  for (int r = 0; r < 8; ++r) {
    int o = o0 + r;
    float g  = ag[r] + baseg[b * Cout + o];
    float bt = ab[r] + baseb[b * Cout + o];
    float bn = (h2f(av[r]) - mean[o]) * rstd[o] * bnw[o] + bnb[o];
    dv[r] = f2h(fmaxf(g * bn + bt, 0.f));
  }
  stv8(outp + ((size_t)b * N + p) * Cout + o0, dv);
}

// ---------------------------------------------------------------------------
// K9: conv1 3x3 over nearest-2x-upsampled film1 (implicit GEMM, WMMA)
//     src CL [b][1024][IN], dst CL [b][4096][OUT]
// ---------------------------------------------------------------------------
__global__ void k_conv1_up(const bf16_t* __restrict__ Wpk, const bf16_t* __restrict__ src,
                           bf16_t* __restrict__ dst) {
  int b = blockIdx.z, ot = blockIdx.y, pt = blockIdx.x * 16;
  int lane = threadIdx.x, m = lane & 15, khalf = lane >> 4;
  int p = pt + m, y = p >> 6, x = p & 63;
  const int Kt = cIN >> 5;                        // 16
  const int fragsPerTap = (cOUT / 16) * Kt * 32;  // v16bf units
  const bf16_t* src_b = src + (size_t)b * cP1 * cIN;
  const v16bf zero = {};
  v8f acc = {};
  for (int tap = 0; tap < 9; ++tap) {
    int ky = tap / 3;
    int yy = y + ky - 1, xx = x + (tap - ky * 3) - 1;
    bool ok = (yy >= 0) & (yy < 64) & (xx >= 0) & (xx < 64);
    int sp = ok ? ((yy >> 1) * 32 + (xx >> 1)) : 0;
    const bf16_t* brow = src_b + (size_t)sp * cIN + (khalf << 4);
    const v16bf* ap = reinterpret_cast<const v16bf*>(Wpk) + (size_t)tap * fragsPerTap +
                      (size_t)ot * Kt * 32 + lane;
    for (int kt = 0; kt < Kt; ++kt) {
      if (kt + 1 < Kt) __builtin_prefetch(brow + (size_t)(kt + 1) * 32, 0, 1);
      v16bf bb = ldv16(brow + (size_t)kt * 32);
      if (!ok) bb = zero;
      acc = wmma_bf16(ap[(size_t)kt * 32], bb, acc);
    }
  }
  stv8(dst + ((size_t)b * cP2 + p) * cOUT + ot * 16 + (khalf << 3), pack_acc(acc));
}

// ---------------------------------------------------------------------------
// K10: conv2 3x3 (pad 1) + fused shortcut 1x1 on upsampled x + bias -> f32 NCHW
// ---------------------------------------------------------------------------
__global__ void k_conv2_sc(const bf16_t* __restrict__ Wpk, const bf16_t* __restrict__ src,
                           const bf16_t* __restrict__ Spk, const float* __restrict__ bsc,
                           const bf16_t* __restrict__ xb, float* __restrict__ dst) {
  int b = blockIdx.z, ot = blockIdx.y, pt = blockIdx.x * 16;
  int lane = threadIdx.x, m = lane & 15, khalf = lane >> 4;
  int p = pt + m, y = p >> 6, x = p & 63;
  const int Kt = cOUT >> 5;                       // 8
  const int fragsPerTap = (cOUT / 16) * Kt * 32;
  const bf16_t* src_b = src + (size_t)b * cP2 * cOUT;
  const v16bf zero = {};
  v8f acc = {};
  for (int tap = 0; tap < 9; ++tap) {
    int ky = tap / 3;
    int yy = y + ky - 1, xx = x + (tap - ky * 3) - 1;
    bool ok = (yy >= 0) & (yy < 64) & (xx >= 0) & (xx < 64);
    int sp = ok ? (yy * 64 + xx) : 0;
    const bf16_t* brow = src_b + (size_t)sp * cOUT + (khalf << 4);
    const v16bf* ap = reinterpret_cast<const v16bf*>(Wpk) + (size_t)tap * fragsPerTap +
                      (size_t)ot * Kt * 32 + lane;
    for (int kt = 0; kt < Kt; ++kt) {
      v16bf bb = ldv16(brow + (size_t)kt * 32);
      if (!ok) bb = zero;
      acc = wmma_bf16(ap[(size_t)kt * 32], bb, acc);
    }
  }
  // shortcut: 1x1 conv over nearest-2x-upsampled x (channels-last, contiguous)
  {
    const int Kt2 = cIN >> 5;                     // 16
    int sp0 = (y >> 1) * 32 + (x >> 1);
    const bf16_t* xrow = xb + ((size_t)b * cP1 + sp0) * cIN + (khalf << 4);
    const v16bf* sp = reinterpret_cast<const v16bf*>(Spk) + (size_t)ot * Kt2 * 32 + lane;
    for (int kt = 0; kt < Kt2; ++kt) {
      if (kt + 1 < Kt2) __builtin_prefetch(xrow + (size_t)(kt + 1) * 32, 0, 1);
      acc = wmma_bf16(sp[(size_t)kt * 32], ldv16(xrow + (size_t)kt * 32), acc);
    }
  }
  int o0 = ot * 16 + (khalf << 3);
#pragma unroll
  for (int r = 0; r < 8; ++r) {
    int o = o0 + r;
    dst[((size_t)b * cOUT + o) * cP2 + p] = acc[r] + bsc[o];
  }
}

// ---------------------------------------------------------------------------
// Host launch
// ---------------------------------------------------------------------------
extern "C" void kernel_launch(void* const* d_in, const int* in_sizes, int n_in,
                              void* d_out, int out_size, void* d_ws, size_t ws_size,
                              hipStream_t stream) {
  const float* x      = (const float*)d_in[0];
  const float* gcond  = (const float*)d_in[1];
  const float* words  = (const float*)d_in[2];
  const unsigned char* mask = (const unsigned char*)d_in[3];
  const float* w_img1 = (const float*)d_in[4];
  const float* w_img2 = (const float*)d_in[5];
  const float* w_c1   = (const float*)d_in[6];
  const float* w_c2   = (const float*)d_in[7];
  const float* bn1w   = (const float*)d_in[8];
  const float* bn1b   = (const float*)d_in[9];
  const float* bn2w   = (const float*)d_in[10];
  const float* bn2b   = (const float*)d_in[11];
  const float* w_g1   = (const float*)d_in[12];
  const float* b_g1   = (const float*)d_in[13];
  const float* w_b1   = (const float*)d_in[14];
  const float* b_b1   = (const float*)d_in[15];
  const float* w_g2   = (const float*)d_in[16];
  const float* b_g2   = (const float*)d_in[17];
  const float* w_b2   = (const float*)d_in[18];
  const float* b_b2   = (const float*)d_in[19];
  const float* w_sc   = (const float*)d_in[20];
  const float* b_sc   = (const float*)d_in[21];
  float* outp = (float*)d_out;

  char* ws = (char*)d_ws;
  size_t off = 0;
  auto alloc = [&](size_t bytes) -> void* {
    off = (off + 255) & ~(size_t)255;
    void* p = ws + off;
    off += bytes;
    return p;
  };

  // activations (channels-last bf16)
  bf16_t* xb       = (bf16_t*)alloc((size_t)cB * cP1 * cIN * 2);
  bf16_t* wlc      = (bf16_t*)alloc((size_t)cB * cTD * cL * 2);
  bf16_t* wcl      = (bf16_t*)alloc((size_t)cB * cL * cTD * 2);
  bf16_t* imgbuf   = (bf16_t*)alloc((size_t)cB * cP2 * cTD * 2);
  bf16_t* ctxbuf   = (bf16_t*)alloc((size_t)cB * cP2 * cTD * 2);
  bf16_t* filmbuf  = (bf16_t*)alloc((size_t)cB * cP2 * cOUT * 2);
  bf16_t* conv1out = (bf16_t*)alloc((size_t)cB * cP2 * cOUT * 2);
  // fragment-packed weights
  bf16_t* pk_img1 = (bf16_t*)alloc((size_t)(cTD/16)*(cIN/32)*512 * 2);
  bf16_t* pk_img2 = (bf16_t*)alloc((size_t)(cTD/16)*(cOUT/32)*512 * 2);
  bf16_t* pk_g1   = (bf16_t*)alloc((size_t)(cIN/16)*(cTD/32)*512 * 2);
  bf16_t* pk_b1   = (bf16_t*)alloc((size_t)(cIN/16)*(cTD/32)*512 * 2);
  bf16_t* pk_g2   = (bf16_t*)alloc((size_t)(cOUT/16)*(cTD/32)*512 * 2);
  bf16_t* pk_b2   = (bf16_t*)alloc((size_t)(cOUT/16)*(cTD/32)*512 * 2);
  bf16_t* pk_c1   = (bf16_t*)alloc((size_t)9*(cOUT/16)*(cIN/32)*512 * 2);
  bf16_t* pk_c2   = (bf16_t*)alloc((size_t)9*(cOUT/16)*(cOUT/32)*512 * 2);
  bf16_t* pk_sc   = (bf16_t*)alloc((size_t)(cOUT/16)*(cIN/32)*512 * 2);
  // scalars
  float* mean1  = (float*)alloc(cIN * 4);
  float* rstd1  = (float*)alloc(cIN * 4);
  float* mean2  = (float*)alloc(cOUT * 4);
  float* rstd2  = (float*)alloc(cOUT * 4);
  float* base1g = (float*)alloc((size_t)cB * cIN * 4);
  float* base1b = (float*)alloc((size_t)cB * cIN * 4);
  float* base2g = (float*)alloc((size_t)cB * cOUT * 4);
  float* base2b = (float*)alloc((size_t)cB * cOUT * 4);

  auto pack = [&](const float* src, bf16_t* dst, int M, int K, int rowPitch,
                  int colOff, int kStride, int taps) {
    int total = (M / 16) * (K / 32) * 32;
    k_pack_a<<<dim3((total + 255) / 256, taps), 256, 0, stream>>>(src, dst, M, K,
                                                                  rowPitch, colOff, kStride);
  };

  // ---- weight packing + prologue ----
  pack(w_img1, pk_img1, cTD, cIN,  cIN,     0,   1, 1);
  pack(w_img2, pk_img2, cTD, cOUT, cOUT,    0,   1, 1);
  pack(w_g1,   pk_g1,   cIN, cTD,  cCOND,   cGD, 1, 1);
  pack(w_b1,   pk_b1,   cIN, cTD,  cCOND,   cGD, 1, 1);
  pack(w_g2,   pk_g2,   cOUT, cTD, cCOND,   cGD, 1, 1);
  pack(w_b2,   pk_b2,   cOUT, cTD, cCOND,   cGD, 1, 1);
  pack(w_c1,   pk_c1,   cOUT, cIN,  cIN * 9,  0, 9, 9);
  pack(w_c2,   pk_c2,   cOUT, cOUT, cOUT * 9, 0, 9, 9);
  pack(w_sc,   pk_sc,   cOUT, cIN,  cIN,      0, 1, 1);

  {
    int n = cB * cIN * cP1;
    k_cast_cl<<<(n + 255) / 256, 256, 0, stream>>>(x, xb, cIN, cP1);
  }
  k_bn_stats_cl<<<cIN, 256, 0, stream>>>(xb, mean1, rstd1, cIN, cP1);
  k_words_norm<<<(cB * cL + 255) / 256, 256, 0, stream>>>(words, wlc, wcl);

  // ---- stage 1 ----
  k_gemm_embed<<<dim3(cP1 / 16, cTD / 16, cB), 32, 0, stream>>>(pk_img1, xb, imgbuf,
                                                                cTD, cIN, cP1);
  k_colnorm_cl<<<(cB * cP1 + 255) / 256, 256, 0, stream>>>(imgbuf, cP1);
  k_context<<<dim3(cP1 / 16, cB), 32, 0, stream>>>(imgbuf, wcl, wlc, mask, ctxbuf, cP1);
  k_cond_base<<<(cB * cIN + 255) / 256, 256, 0, stream>>>(w_g1, b_g1, w_b1, b_b1, gcond,
                                                          base1g, base1b, cIN);
  k_film<<<dim3(cP1 / 16, cIN / 16, cB), 32, 0, stream>>>(
      pk_g1, pk_b1, base1g, base1b, ctxbuf, xb, mean1, rstd1, bn1w, bn1b,
      filmbuf, cIN, cP1);
  k_conv1_up<<<dim3(cP2 / 16, cOUT / 16, cB), 32, 0, stream>>>(pk_c1, filmbuf, conv1out);

  // ---- stage 2 ----
  k_bn_stats_cl<<<cOUT, 256, 0, stream>>>(conv1out, mean2, rstd2, cOUT, cP2);
  k_gemm_embed<<<dim3(cP2 / 16, cTD / 16, cB), 32, 0, stream>>>(pk_img2, conv1out, imgbuf,
                                                                cTD, cOUT, cP2);
  k_colnorm_cl<<<(cB * cP2 + 255) / 256, 256, 0, stream>>>(imgbuf, cP2);
  k_context<<<dim3(cP2 / 16, cB), 32, 0, stream>>>(imgbuf, wcl, wlc, mask, ctxbuf, cP2);
  k_cond_base<<<(cB * cOUT + 255) / 256, 256, 0, stream>>>(w_g2, b_g2, w_b2, b_b2, gcond,
                                                           base2g, base2b, cOUT);
  k_film<<<dim3(cP2 / 16, cOUT / 16, cB), 32, 0, stream>>>(
      pk_g2, pk_b2, base2g, base2b, ctxbuf, conv1out, mean2, rstd2, bn2w, bn2b,
      filmbuf, cOUT, cP2);
  k_conv2_sc<<<dim3(cP2 / 16, cOUT / 16, cB), 32, 0, stream>>>(pk_c2, filmbuf, pk_sc, b_sc,
                                                               xb, outp);

  (void)in_sizes; (void)n_in; (void)out_size; (void)ws_size;
}
